// LocalEncoder_12549894439060
// MI455X (gfx1250) — matrix operand order repaired and checked by reading, not compiled, not run on hardware
//
#include <hip/hip_runtime.h>
#include <hip/hip_bf16.h>

// ---------------------------------------------------------------------------
// LocalEncoder on MI455X (gfx1250, wave32).
// Dense GEMMs + self-attention on v_wmma_f32_16x16x32_f16 (f32 accum).
// GEMM weight tiles streamed with GLOBAL_LOAD_ASYNC_TO_LDS_B128 (ASYNCcnt).
// Self-attention fused flash-style (no B*H*T*T score tensor).
// Cross-attention exploits contiguity of patches (cumsum ids) -> sparse kernel.
// Layer-0 cross path skipped (reference discards it).
// ---------------------------------------------------------------------------

#define D_    384
#define T_    2048
#define B_    4
#define H_    8
#define HD_   48
#define FFD_  1536
#define V_    100000

typedef _Float16 h16;
typedef __attribute__((ext_vector_type(16))) _Float16    v16h;
typedef __attribute__((ext_vector_type(8)))  float       v8f;
typedef __attribute__((ext_vector_type(8)))  unsigned int v8u;
typedef unsigned int u32;
typedef unsigned long long u64;

// ---------------- WMMA helpers (wave32, 16x16x32 f16 -> f32) ----------------

__device__ inline v8f wmma_f16(v16h a, v16h b, v8f c) {
  return __builtin_amdgcn_wmma_f32_16x16x32_f16(
      /*neg_a=*/false, a, /*neg_b=*/false, b,
      /*c_mod=*/(short)0, c, /*reuse_a=*/false, /*reuse_b=*/false);
}

// A fragment: 16x32 f16 tile, row-major in LDS with row stride `ld` halves.
// Layout: lanes0-15 (M=lane) K=0..15; lanes16-31 same M, K=16..31 (matches B).
__device__ inline v16h lda_frag(const h16* p, int ld, int lane) {
  int m = lane & 15, hi = (lane >> 4) & 1;
  const u32* q = (const u32*)(p + m * ld + hi * 16);
  v8u u;
#pragma unroll
  for (int r = 0; r < 8; ++r) u[r] = q[r];
  return __builtin_bit_cast(v16h, u);
}

// B fragment: 32x16 f16; LDS holds B^T row-major [N][K] (stride `ld` halves).
// lanes0-15: N=lane, K=0..15; lanes16-31: same N, K=16..31 -> contiguous 32B.
__device__ inline v16h ldb_frag(const h16* p, int ld, int lane) {
  int n = lane & 15, hi = (lane >> 4) & 1;
  const u32* q = (const u32*)(p + n * ld + hi * 16);
  v8u u;
#pragma unroll
  for (int r = 0; r < 8; ++r) u[r] = q[r];
  return __builtin_bit_cast(v16h, u);
}

// ---------------- block reductions (128 threads = 4 waves) ------------------

__device__ inline float block_sum128(float v, float* sbuf, int tid) {
#pragma unroll
  for (int o = 16; o >= 1; o >>= 1) v += __shfl_xor(v, o, 32);
  if ((tid & 31) == 0) sbuf[tid >> 5] = v;
  __syncthreads();
  v = sbuf[0] + sbuf[1] + sbuf[2] + sbuf[3];
  __syncthreads();
  return v;
}
__device__ inline float block_max128(float v, float* sbuf, int tid) {
#pragma unroll
  for (int o = 16; o >= 1; o >>= 1) v = fmaxf(v, __shfl_xor(v, o, 32));
  if ((tid & 31) == 0) sbuf[tid >> 5] = v;
  __syncthreads();
  v = fmaxf(fmaxf(sbuf[0], sbuf[1]), fmaxf(sbuf[2], sbuf[3]));
  __syncthreads();
  return v;
}

// ---------------- embedding + ngram hash ------------------------------------

__global__ __launch_bounds__(128) void embed_kernel(
    const int* __restrict__ ids, const float* __restrict__ BE,
    const float* __restrict__ T3, const float* __restrict__ T4,
    const float* __restrict__ T5, float* __restrict__ X) {
  int bt = blockIdx.x;
  int b = bt / T_, t = bt % T_;
  const int* row = ids + (size_t)b * T_;
  int id = row[t];
  int h3 = 0, h4 = 0, h5 = 0;
  bool v3 = (t >= 2), v4 = (t >= 3), v5 = (t >= 4);
  if (v3) { int h = 0; for (int j = t - 2; j <= t; ++j) h = (h * 31 + row[j]) % V_; h3 = h; }
  if (v4) { int h = 0; for (int j = t - 3; j <= t; ++j) h = (h * 31 + row[j]) % V_; h4 = h; }
  if (v5) { int h = 0; for (int j = t - 4; j <= t; ++j) h = (h * 31 + row[j]) % V_; h5 = h; }
  size_t ob = (size_t)bt * D_;
  for (int d = threadIdx.x; d < D_; d += 128) {
    float s = 0.f;
    if (v3) s += T3[(size_t)h3 * D_ + d];
    if (v4) s += T4[(size_t)h4 * D_ + d];
    if (v5) s += T5[(size_t)h5 * D_ + d];
    X[ob + d] = BE[(size_t)id * D_ + d] + 0.25f * s;
  }
}

// ---------------- patch segmentation (cumsum -> contiguous ranges) ----------

__global__ void patch_kernel(const int* __restrict__ pb,
                             int* __restrict__ starts, int* __restrict__ counts) {
  int b = threadIdx.x;
  if (b >= B_) return;
  int run = 0;
  for (int t = 0; t < T_; ++t) {
    if (pb[b * T_ + t] == 1) { starts[b * (T_ + 1) + run] = t; run++; }
  }
  counts[b] = run;
  starts[b * (T_ + 1) + run] = T_;
}

// ---------------- weight transpose + f16 convert ----------------------------

__global__ void convert_wT(const float* __restrict__ W, h16* __restrict__ Bt,
                           int Kd, int N) {
  int i = blockIdx.x * 256 + threadIdx.x;
  if (i >= N * Kd) return;
  int n = i / Kd, k = i % Kd;
  Bt[i] = (h16)W[(size_t)k * N + n];
}

// ---------------- generic WMMA GEMM: C = A[MxK] * W[KxN] (+bias,res,relu) ---
// A: f32 global (float4 loads, f16 convert in LDS stage).
// Bt: f16 global [N][K], streamed via GLOBAL_LOAD_ASYNC_TO_LDS_B128.
// 128 threads = 4 waves; 128x64 tile; each wave owns 32 rows:
//   2 A-frags reuse the same 4 B-frags -> 8 WMMAs / k-step / wave.

__global__ __launch_bounds__(128) void gemm_kernel(
    const float* __restrict__ A, const h16* __restrict__ Bt,
    const float* __restrict__ bias, const float* __restrict__ residual,
    float* __restrict__ C, int M, int N, int Kd, int flags) {
  __shared__ __align__(16) h16 As[128][40];   // 80B row pitch (16B multiple)
  __shared__ __align__(16) h16 Bs[64][40];
  int tid = threadIdx.x, lane = tid & 31, w = tid >> 5;
  int bm = blockIdx.y * 128, bn = blockIdx.x * 64;
  v8f zero8 = {0.f, 0.f, 0.f, 0.f, 0.f, 0.f, 0.f, 0.f};
  v8f acc[2][4];
#pragma unroll
  for (int mt = 0; mt < 2; ++mt)
#pragma unroll
    for (int nt = 0; nt < 4; ++nt) acc[mt][nt] = zero8;

  for (int k0 = 0; k0 < Kd; k0 += 32) {
    __syncthreads();
    // ---- Bt tile (64 rows x 32 halves): async DMA global -> LDS, 16B/lane.
    // Row = 64B of payload = 4 chunks; 256 chunks over 128 threads.
#pragma unroll
    for (int it = 0; it < 2; ++it) {
      int i = tid + it * 128;
      int r = i >> 2, c = i & 3;
      u32 lds = (u32)(uintptr_t)&Bs[r][c * 8];
      u64 g = (u64)(uintptr_t)(Bt + (size_t)(bn + r) * Kd + k0 + c * 8);
      asm volatile("global_load_async_to_lds_b128 %0, %1, off"
                   :: "v"(lds), "v"(g) : "memory");
    }
    // ---- A tile (128 rows x 32 f32): float4 loads, cvt, packed b64 stores.
#pragma unroll
    for (int it = 0; it < 8; ++it) {
      int i = tid + it * 128;
      int r = i >> 3, c4 = (i & 7) * 4;
      int gr = bm + r;
      float4 v = make_float4(0.f, 0.f, 0.f, 0.f);
      if (gr < M) v = *(const float4*)(A + (size_t)gr * Kd + k0 + c4);
      union { h16 h[4]; u64 q; } pk;
      pk.h[0] = (h16)v.x; pk.h[1] = (h16)v.y;
      pk.h[2] = (h16)v.z; pk.h[3] = (h16)v.w;
      *(u64*)&As[r][c4] = pk.q;
    }
    asm volatile("s_wait_asynccnt 0" ::: "memory");
    __syncthreads();

    v16h a0 = lda_frag(&As[w * 32][0], 40, lane);
    v16h a1 = lda_frag(&As[w * 32 + 16][0], 40, lane);
#pragma unroll
    for (int nt = 0; nt < 4; ++nt) {
      v16h bf = ldb_frag(&Bs[nt * 16][0], 40, lane);
      acc[0][nt] = wmma_f16(a0, bf, acc[0][nt]);
      acc[1][nt] = wmma_f16(a1, bf, acc[1][nt]);
    }
  }
  int hi = lane >> 4, nn = lane & 15;
#pragma unroll
  for (int mt = 0; mt < 2; ++mt)
#pragma unroll
    for (int nt = 0; nt < 4; ++nt) {
      int col = bn + nt * 16 + nn;
      float bv = bias ? bias[col] : 0.f;
#pragma unroll
      for (int r = 0; r < 8; ++r) {
        int row = bm + w * 32 + mt * 16 + hi * 8 + r;
        if (row < M) {
          float v = acc[mt][nt][r] + bv;
          if (residual) v += residual[(size_t)row * N + col];
          if (flags & 1) v = fmaxf(v, 0.f);
          C[(size_t)row * N + col] = v;
        }
      }
    }
}

// ---------------- fused flash self-attention (per b,h; 64-query tiles) ------

__global__ __launch_bounds__(128) void attn_kernel(
    const float* __restrict__ Q, const float* __restrict__ Kg,
    const float* __restrict__ Vg, float* __restrict__ O) {
  __shared__ __align__(16) h16 Qs[64][72];
  __shared__ __align__(16) h16 Ks[64][72];
  __shared__ __align__(16) h16 Vt[48][72];        // V^T: [dim][key]
  __shared__ __align__(16) h16 Ps[4][16][72];     // per-wave probs staging
  int tid = threadIdx.x, lane = tid & 31, w = tid >> 5;
  int bh = blockIdx.y, b = bh >> 3, h = bh & 7;
  int q0 = blockIdx.x * 64;
  const float scale = 0.14433756729740643f;       // rsqrt(48)
  size_t base = ((size_t)b * T_) * D_ + h * HD_;

  for (int i = tid; i < 64 * 64; i += 128) {
    int r = i >> 6, c = i & 63;
    float v = (c < HD_) ? Q[base + (size_t)(q0 + r) * D_ + c] * scale : 0.f;
    Qs[r][c] = (h16)v;
  }
  __syncthreads();

  v8f zero8 = {0.f, 0.f, 0.f, 0.f, 0.f, 0.f, 0.f, 0.f};
  v8f oacc[3];
#pragma unroll
  for (int nt = 0; nt < 3; ++nt) oacc[nt] = zero8;
  float mrun[8], lrun[8];
#pragma unroll
  for (int r = 0; r < 8; ++r) { mrun[r] = -3.0e38f; lrun[r] = 0.f; }

  int hi = lane >> 4, nn = lane & 15;

  for (int kt = 0; kt < T_; kt += 64) {
    __syncthreads();
    for (int i = tid; i < 64 * 64; i += 128) {
      int r = i >> 6, c = i & 63;
      float kv = (c < HD_) ? Kg[base + (size_t)(kt + r) * D_ + c] : 0.f;
      Ks[r][c] = (h16)kv;
    }
    for (int i = tid; i < 64 * HD_; i += 128) {
      int r = i / HD_, c = i % HD_;
      Vt[c][r] = (h16)Vg[base + (size_t)(kt + r) * D_ + c];
    }
    __syncthreads();

    // S = Q K^T : 4 key n-tiles x 2 k-steps
    v16h a0 = lda_frag(&Qs[w * 16][0], 72, lane);
    v16h a1 = lda_frag(&Qs[w * 16][32], 72, lane);
    v8f sacc[4];
#pragma unroll
    for (int nt = 0; nt < 4; ++nt) {
      sacc[nt] = zero8;
      v16h b0 = ldb_frag(&Ks[nt * 16][0], 72, lane);
      sacc[nt] = wmma_f16(a0, b0, sacc[nt]);
      v16h b1 = ldb_frag(&Ks[nt * 16][32], 72, lane);
      sacc[nt] = wmma_f16(a1, b1, sacc[nt]);
    }

    // online softmax stats (rows live in 16-lane half-wave groups)
    float mnew[8], f[8], rs[8];
#pragma unroll
    for (int r = 0; r < 8; ++r) {
      float mx = fmaxf(fmaxf(sacc[0][r], sacc[1][r]),
                       fmaxf(sacc[2][r], sacc[3][r]));
#pragma unroll
      for (int o = 8; o >= 1; o >>= 1) mx = fmaxf(mx, __shfl_xor(mx, o, 32));
      mnew[r] = fmaxf(mrun[r], mx);
      f[r] = __expf(mrun[r] - mnew[r]);
      rs[r] = 0.f;
    }
#pragma unroll
    for (int nt = 0; nt < 4; ++nt) {
#pragma unroll
      for (int r = 0; r < 8; ++r) {
        float e = __expf(sacc[nt][r] - mnew[r]);
        rs[r] += e;
        Ps[w][hi * 8 + r][nt * 16 + nn] = (h16)e;   // C-layout -> LDS
      }
    }
#pragma unroll
    for (int r = 0; r < 8; ++r) {
      float s = rs[r];
#pragma unroll
      for (int o = 8; o >= 1; o >>= 1) s += __shfl_xor(s, o, 32);
      lrun[r] = lrun[r] * f[r] + s;
      mrun[r] = mnew[r];
    }
#pragma unroll
    for (int nt = 0; nt < 3; ++nt)
#pragma unroll
      for (int r = 0; r < 8; ++r) oacc[nt][r] *= f[r];

    // wave-private LDS RAW: DS ops in-order per wave; fence the counter.
    asm volatile("s_wait_dscnt 0" ::: "memory");

    // O += P V : reload P in A-layout, V^T gives contiguous B frags
    v16h p0 = lda_frag(&Ps[w][0][0], 72, lane);
    v16h p1 = lda_frag(&Ps[w][0][32], 72, lane);
#pragma unroll
    for (int nt = 0; nt < 3; ++nt) {
      v16h vb0 = ldb_frag(&Vt[nt * 16][0], 72, lane);
      oacc[nt] = wmma_f16(p0, vb0, oacc[nt]);
      v16h vb1 = ldb_frag(&Vt[nt * 16][32], 72, lane);
      oacc[nt] = wmma_f16(p1, vb1, oacc[nt]);
    }
  }

#pragma unroll
  for (int r = 0; r < 8; ++r) {
    float inv = 1.f / lrun[r];
    int row = q0 + w * 16 + hi * 8 + r;
#pragma unroll
    for (int nt = 0; nt < 3; ++nt)
      O[base + (size_t)row * D_ + nt * 16 + nn] = oacc[nt][r] * inv;
  }
}

// ---------------- layernorm (row = 384) -------------------------------------

__global__ __launch_bounds__(128) void ln_kernel(
    const float* __restrict__ X, const float* __restrict__ g,
    const float* __restrict__ bta, float* __restrict__ Y, int rows) {
  __shared__ float sbuf[4];
  int row = blockIdx.x;
  if (row >= rows) return;
  const float* xp = X + (size_t)row * D_;
  int tid = threadIdx.x;
  float v[3], s = 0.f;
#pragma unroll
  for (int i = 0; i < 3; ++i) { v[i] = xp[tid + 128 * i]; s += v[i]; }
  s = block_sum128(s, sbuf, tid);
  float mu = s * (1.f / D_);
  float q = 0.f;
#pragma unroll
  for (int i = 0; i < 3; ++i) { float d = v[i] - mu; q += d * d; }
  q = block_sum128(q, sbuf, tid);
  float inv = rsqrtf(q * (1.f / D_) + 1e-5f);
#pragma unroll
  for (int i = 0; i < 3; ++i) {
    int d = tid + 128 * i;
    Y[(size_t)row * D_ + d] = (v[i] - mu) * inv * g[d] + bta[d];
  }
}

// ---------------- segment-max pooling ---------------------------------------

__global__ __launch_bounds__(128) void pool_kernel(
    const float* __restrict__ X, const int* __restrict__ starts,
    const int* __restrict__ counts, float* __restrict__ QRY, int P) {
  int bp = blockIdx.x, b = bp / P, p = bp % P;
  int cnt = counts[b];
  for (int d = threadIdx.x; d < D_; d += 128) {
    float m = 0.f;                                  // empty patch -> 0
    if (p < cnt) {
      int s = starts[b * (T_ + 1) + p], e = starts[b * (T_ + 1) + p + 1];
      m = -3.4e38f;
      for (int t = s; t < e; ++t) m = fmaxf(m, X[((size_t)b * T_ + t) * D_ + d]);
    }
    QRY[((size_t)b * P + p) * D_ + d] = m;
  }
}

// ---------------- sparse cross-attention (contiguous patch segments) --------
// Empty patch: mask all -1e9 -> softmax uniform over T -> scores := 0 over T.

__global__ __launch_bounds__(128) void cross_attn_kernel(
    const float* __restrict__ Qc, const float* __restrict__ Kc,
    const float* __restrict__ Vc, const int* __restrict__ starts,
    const int* __restrict__ counts, float* __restrict__ CO, int P) {
  __shared__ float sc[T_];
  __shared__ float sbuf[4];
  int bp = blockIdx.x, b = bp / P, p = bp % P;
  int tid = threadIdx.x;
  int cnt = counts[b];
  bool empty = (p >= cnt);
  int s0 = empty ? 0 : starts[b * (T_ + 1) + p];
  int e0 = empty ? T_ : starts[b * (T_ + 1) + p + 1];
  int Ln = e0 - s0;
  const float scale = 0.14433756729740643f;
  for (int h = 0; h < H_; ++h) {
    const float* qp = Qc + ((size_t)b * P + p) * D_ + h * HD_;
    for (int i = tid; i < Ln; i += 128) {
      float acc = 0.f;
      if (!empty) {
        const float* kp = Kc + ((size_t)b * T_ + s0 + i) * D_ + h * HD_;
#pragma unroll
        for (int d = 0; d < HD_; ++d) acc += qp[d] * kp[d];
        acc *= scale;
      }
      sc[i] = acc;
    }
    __syncthreads();
    float mx = -3.0e38f;
    for (int i = tid; i < Ln; i += 128) mx = fmaxf(mx, sc[i]);
    mx = block_max128(mx, sbuf, tid);
    float zs = 0.f;
    for (int i = tid; i < Ln; i += 128) zs += __expf(sc[i] - mx);
    zs = block_sum128(zs, sbuf, tid);
    if (tid < HD_) {
      float o = 0.f;
      for (int i = 0; i < Ln; ++i)
        o += __expf(sc[i] - mx) * Vc[((size_t)b * T_ + s0 + i) * D_ + h * HD_ + tid];
      CO[((size_t)b * P + p) * D_ + h * HD_ + tid] = o / zs;
    }
    __syncthreads();
  }
}

// ---------------- output tiling (K=2) ---------------------------------------

__global__ void tile_out_kernel(const float* __restrict__ PR,
                                float* __restrict__ out, int n) {
  int i = blockIdx.x * 256 + threadIdx.x;
  if (i >= n) return;
  float v = PR[i];
  int row = i / D_, d = i % D_;
  out[(size_t)row * (2 * D_) + d] = v;
  out[(size_t)row * (2 * D_) + D_ + d] = v;
}

// ---------------- host launcher ---------------------------------------------

extern "C" void kernel_launch(void* const* d_in, const int* in_sizes, int n_in,
                              void* d_out, int out_size, void* d_ws, size_t ws_size,
                              hipStream_t stream) {
  const int P = out_size / (B_ * D_ * 2);
  const bool ins = (n_in > 1 && in_sizes[1] == B_ * T_);  // insertion-order vs sorted pytree
  auto F = [&](int i) -> const float* { return (const float*)d_in[i]; };

  const int* ids = (const int*)d_in[0];
  const int* pb = (const int*)d_in[ins ? 1 : 65];
  const float* BE = F(ins ? 2 : 1);
  const float *T3, *T4, *T5;
  if (ins) { T3 = F(3); T4 = F(4); T5 = F(5); }
  else     { T3 = F(62); T4 = F(63); T5 = F(64); }

  struct LW {
    const float *wq, *wk, *wv, *wo, *bq, *bk, *bv, *bo, *w1, *b1, *w2, *b2,
        *ln1g, *ln1b, *ln2g, *ln2b, *cwq, *cwk, *cwv, *cwo, *cbq, *cbk, *cbv,
        *cbo, *nqg, *nqb, *nkvg, *nkvb, *ng, *nb;
  } lw[2];
  for (int l = 0; l < 2; ++l) {
    LW& L = lw[l];
    if (ins) {
      int s = 6 + 30 * l;
      L.wq = F(s + 0);  L.wk = F(s + 1);  L.wv = F(s + 2);  L.wo = F(s + 3);
      L.bq = F(s + 4);  L.bk = F(s + 5);  L.bv = F(s + 6);  L.bo = F(s + 7);
      L.w1 = F(s + 8);  L.b1 = F(s + 9);  L.w2 = F(s + 10); L.b2 = F(s + 11);
      L.ln1g = F(s + 12); L.ln1b = F(s + 13); L.ln2g = F(s + 14); L.ln2b = F(s + 15);
      L.cwq = F(s + 16); L.cwk = F(s + 17); L.cwv = F(s + 18); L.cwo = F(s + 19);
      L.cbq = F(s + 20); L.cbk = F(s + 21); L.cbv = F(s + 22); L.cbo = F(s + 23);
      L.nqg = F(s + 24); L.nqb = F(s + 25); L.nkvg = F(s + 26); L.nkvb = F(s + 27);
      L.ng = F(s + 28); L.nb = F(s + 29);
    } else {                       // alphabetical pytree flatten
      int s = 2 + 30 * l;
      L.cbk = F(s + 0);  L.cbo = F(s + 1);  L.cbq = F(s + 2);  L.cbv = F(s + 3);
      L.nkvb = F(s + 4); L.nkvg = F(s + 5); L.nqb = F(s + 6);  L.nqg = F(s + 7);
      L.cwk = F(s + 8);  L.cwo = F(s + 9);  L.cwq = F(s + 10); L.cwv = F(s + 11);
      L.nb = F(s + 12);  L.ng = F(s + 13);
      L.b1 = F(s + 14);  L.b2 = F(s + 15);
      L.bk = F(s + 16);  L.bo = F(s + 17);  L.bq = F(s + 18);  L.bv = F(s + 19);
      L.ln1b = F(s + 20); L.ln1g = F(s + 21); L.ln2b = F(s + 22); L.ln2g = F(s + 23);
      L.w1 = F(s + 24);  L.w2 = F(s + 25);
      L.wk = F(s + 26);  L.wo = F(s + 27);  L.wq = F(s + 28);  L.wv = F(s + 29);
    }
  }

  // workspace carve
  char* wp = (char*)d_ws;
  size_t off = 0;
  auto carve = [&](size_t bytes) -> void* {
    void* p = wp + off;
    off = (off + bytes + 255) & ~(size_t)255;
    return p;
  };
  const size_t NT = (size_t)B_ * T_;
  const size_t A34 = NT * D_;
  float* X  = (float*)carve(A34 * 4);
  float* Qb = (float*)carve(A34 * 4);
  float* Kb = (float*)carve(A34 * 4);
  float* Vb = (float*)carve(A34 * 4);
  float* AT = (float*)carve(A34 * 4);
  float* T1 = (float*)carve(A34 * 4);
  float* FF = (float*)carve(NT * FFD_ * 4);
  h16*  WH  = (h16*)carve((size_t)D_ * FFD_ * 2);
  int* starts = (int*)carve(B_ * (T_ + 1) * 4);
  int* counts = (int*)carve(B_ * 4);
  // cross-phase buffers alias FF (free after FFN)
  float* QRY = FF;
  float* QN  = FF + A34;
  float* KVN = FF + 2 * A34;
  float* CO  = FF + 3 * A34;
  (void)ws_size;

  auto conv = [&](const float* W, int Kd, int N) {
    convert_wT<<<(N * Kd + 255) / 256, 256, 0, stream>>>(W, WH, Kd, N);
  };
  auto gemm = [&](const float* A, const float* bias, const float* res,
                  float* Cmat, int M, int N, int Kd, int flags) {
    dim3 g(N / 64, (M + 127) / 128);
    gemm_kernel<<<g, 128, 0, stream>>>(A, WH, bias, res, Cmat, M, N, Kd, flags);
  };
  auto ln = [&](const float* Xi, const float* g_, const float* b_, float* Y,
                int rows) { ln_kernel<<<rows, 128, 0, stream>>>(Xi, g_, b_, Y, rows); };

  // ---- embedding + patches ----
  embed_kernel<<<(int)NT, 128, 0, stream>>>(ids, BE, T3, T4, T5, X);
  patch_kernel<<<1, 32, 0, stream>>>(pb, starts, counts);

  // ---- encoder layers ----
  for (int l = 0; l < 2; ++l) {
    LW& L = lw[l];
    conv(L.wq, D_, D_);  gemm(X, L.bq, nullptr, Qb, (int)NT, D_, D_, 0);
    conv(L.wk, D_, D_);  gemm(X, L.bk, nullptr, Kb, (int)NT, D_, D_, 0);
    conv(L.wv, D_, D_);  gemm(X, L.bv, nullptr, Vb, (int)NT, D_, D_, 0);
    attn_kernel<<<dim3(T_ / 64, B_ * H_), 128, 0, stream>>>(Qb, Kb, Vb, AT);
    conv(L.wo, D_, D_);  gemm(AT, L.bo, X, T1, (int)NT, D_, D_, 0);
    ln(T1, L.ln1g, L.ln1b, X, (int)NT);
    conv(L.w1, D_, FFD_); gemm(X, L.b1, nullptr, FF, (int)NT, FFD_, D_, 1);
    conv(L.w2, FFD_, D_); gemm(FF, L.b2, X, T1, (int)NT, D_, FFD_, 0);
    ln(T1, L.ln2g, L.ln2b, X, (int)NT);
  }

  // ---- cross-attention (final layer only; layer-0 pr is discarded) ----
  LW& L = lw[1];
  pool_kernel<<<B_ * P, 128, 0, stream>>>(X, starts, counts, QRY, P);
  ln(QRY, L.nqg, L.nqb, QN, B_ * P);
  ln(X, L.nkvg, L.nkvb, KVN, (int)NT);
  conv(L.cwq, D_, D_); gemm(QN, L.cbq, nullptr, Qb, B_ * P, D_, D_, 0);
  conv(L.cwk, D_, D_); gemm(KVN, L.cbk, nullptr, Kb, (int)NT, D_, D_, 0);
  conv(L.cwv, D_, D_); gemm(KVN, L.cbv, nullptr, Vb, (int)NT, D_, D_, 0);
  cross_attn_kernel<<<B_ * P, 128, 0, stream>>>(Qb, Kb, Vb, starts, counts, CO, P);
  conv(L.cwo, D_, D_); gemm(CO, L.cbo, QRY, T1, B_ * P, D_, D_, 0);
  ln(T1, L.ng, L.nb, AT, B_ * P);

  tile_out_kernel<<<(B_ * P * D_ + 255) / 256, 256, 0, stream>>>(
      AT, (float*)d_out, B_ * P * D_);
}
